// Attention_31722628448792
// MI455X (gfx1250) — matrix operand order repaired and checked by reading, not compiled, not run on hardware
//
#include <hip/hip_runtime.h>

typedef __bf16 bf16_t;
typedef __attribute__((ext_vector_type(16))) __bf16 v16bf;
typedef __attribute__((ext_vector_type(8)))  __bf16 v8bf;
typedef __attribute__((ext_vector_type(4)))  __bf16 v4bf;
typedef __attribute__((ext_vector_type(8)))  float  v8f;
typedef __attribute__((ext_vector_type(4)))  float  v4f;

static constexpr int BB  = 2;
static constexpr int TT  = 2048;
static constexpr int DD  = 2048;
static constexpr int NH  = 16;
static constexpr int NKV = 4;
static constexpr int HD  = 128;
static constexpr float SCALE = 0.08838834764831845f;   // 1/sqrt(128)

#if defined(__gfx1250__) && __has_builtin(__builtin_amdgcn_tensor_load_to_lds) && __has_builtin(__builtin_amdgcn_s_wait_tensorcnt)
#define HAVE_TDM 1
typedef __attribute__((ext_vector_type(4))) unsigned int u32x4;
typedef __attribute__((ext_vector_type(8))) int          i32x8;
typedef __attribute__((ext_vector_type(4))) int          i32x4;
#else
#define HAVE_TDM 0
#endif

// ---------------------------------------------------------------- WMMA helpers

__device__ inline v8f wmma_bf16(v16bf a, v16bf b, v8f c) {
  // (neg_a, A, neg_b, B, c_mod, C, reuse_a, reuse_b)
  return __builtin_amdgcn_wmma_f32_16x16x32_bf16(false, a, false, b, (short)0, c, false, false);
}

// A operand (16x32, 16-bit): lane half h holds K = {8h..8h+7, 16+8h..16+8h+7}
__device__ inline v16bf frag_a(const bf16_t* p, int hf) {
  v8bf lo = *(const v8bf*)(p + hf * 8);
  v8bf hi = *(const v8bf*)(p + hf * 8 + 16);
  v16bf r;
#pragma unroll
  for (int i = 0; i < 8; ++i) { r[i] = lo[i]; r[i + 8] = hi[i]; }
  return r;
}

// B operand (32x16, 16-bit): lane half h holds K = {8h..8h+7, 16+8h..16+8h+7}
__device__ inline v16bf frag_b(const bf16_t* p, int hf) {
  v8bf lo = *(const v8bf*)(p + hf * 8);
  v8bf hi = *(const v8bf*)(p + hf * 8 + 16);
  v16bf r;
#pragma unroll
  for (int i = 0; i < 8; ++i) { r[i] = lo[i]; r[i + 8] = hi[i]; }
  return r;
}

__device__ inline float wave_sum32(float v) {
#pragma unroll
  for (int off = 16; off > 0; off >>= 1) v += __shfl_xor(v, off, 32);
  return v;
}

// ---------------------------------------------------------------- GEMM (fp32 in, bf16 WMMA, fp32 out)
// C[M,N] = A[M,K] * W[K,N].  Block = 64x64 tile, 128 threads (4 waves, 2x2 of 32x32).

__global__ __launch_bounds__(128) void gemm_bf16w(const float* __restrict__ A,
                                                  const float* __restrict__ W,
                                                  float* __restrict__ C,
                                                  int M, int N, int K) {
  __shared__ bf16_t lA[64 * 32];   // [m][k]
  __shared__ bf16_t lB[64 * 32];   // [n][k]  (W transposed)
  const int tid  = threadIdx.x;
  const int wave = tid >> 5, lane = tid & 31;
  const int hf   = lane >> 4, l16 = lane & 15;
  const int m0 = blockIdx.x * 64, n0 = blockIdx.y * 64;
  const int wm = (wave >> 1) * 32, wn = (wave & 1) * 32;

  const v8f vzero = {0.f, 0.f, 0.f, 0.f, 0.f, 0.f, 0.f, 0.f};
  v8f acc[2][2];
#pragma unroll
  for (int i = 0; i < 2; ++i)
#pragma unroll
    for (int j = 0; j < 2; ++j) acc[i][j] = vzero;

  for (int k0 = 0; k0 < K; k0 += 32) {
    __syncthreads();
    // ---- A tile: 64x32 fp32 -> bf16.  Fixed 4 iterations (no exec masking);
    //      batch all 4 global b128 loads before converting/storing.
    {
      v4f va[4];
#pragma unroll
      for (int it = 0; it < 4; ++it) {
        int e4 = tid + it * 128;
        int m  = e4 >> 3;              // 8 float4 per 32-wide row
        int kk = (e4 & 7) * 4;
        va[it] = *(const v4f*)(A + (size_t)(m0 + m) * K + k0 + kk);
      }
#pragma unroll
      for (int it = 0; it < 4; ++it) {
        int e4 = tid + it * 128;
        int m  = e4 >> 3;
        int kk = (e4 & 7) * 4;
        v4bf t;
        t[0] = (bf16_t)va[it][0]; t[1] = (bf16_t)va[it][1];
        t[2] = (bf16_t)va[it][2]; t[3] = (bf16_t)va[it][3];
        *(v4bf*)(lA + m * 32 + kk) = t;
      }
    }
    // ---- B tile transposed: 4 coalesced W rows -> one 64-bit LDS store each.
    {
      float wbuf[4][4];
#pragma unroll
      for (int it = 0; it < 4; ++it) {
        int e4 = tid + it * 128;
        int n  = e4 & 63;
        int kk = (e4 >> 6) * 4;
        const float* wp = W + (size_t)(k0 + kk) * N + n0 + n;
#pragma unroll
        for (int j = 0; j < 4; ++j) wbuf[it][j] = wp[(size_t)j * N];
      }
#pragma unroll
      for (int it = 0; it < 4; ++it) {
        int e4 = tid + it * 128;
        int n  = e4 & 63;
        int kk = (e4 >> 6) * 4;
        v4bf t;
        t[0] = (bf16_t)wbuf[it][0]; t[1] = (bf16_t)wbuf[it][1];
        t[2] = (bf16_t)wbuf[it][2]; t[3] = (bf16_t)wbuf[it][3];
        *(v4bf*)(lB + n * 32 + kk) = t;
      }
    }
    if (k0 + 32 < K) {   // prefetch next chunk (global_prefetch_b8)
      __builtin_prefetch(A + (size_t)(m0 + (tid >> 1)) * K + k0 + 32 + (tid & 1) * 16, 0, 0);
      __builtin_prefetch(W + (size_t)(k0 + 32 + (tid & 31)) * N + n0 + (tid >> 5) * 16, 0, 0);
    }
    __syncthreads();

    v16bf af[2], bfr[2];
#pragma unroll
    for (int i = 0; i < 2; ++i) af[i] = frag_a(lA + (wm + i * 16 + l16) * 32, hf);
#pragma unroll
    for (int j = 0; j < 2; ++j) bfr[j] = frag_b(lB + (wn + j * 16 + l16) * 32, hf);
#pragma unroll
    for (int i = 0; i < 2; ++i)
#pragma unroll
      for (int j = 0; j < 2; ++j) acc[i][j] = wmma_bf16(af[i], bfr[j], acc[i][j]);
  }

#pragma unroll
  for (int i = 0; i < 2; ++i)
#pragma unroll
    for (int j = 0; j < 2; ++j)
#pragma unroll
      for (int r = 0; r < 8; ++r) {
        int m = m0 + wm + i * 16 + hf * 8 + r;
        int n = n0 + wn + j * 16 + l16;
        C[(size_t)m * N + n] = acc[i][j][r];
      }
}

// ---------------------------------------------------------------- pack Q: rmsnorm + rope -> bf16 [B,NH,T,HD]
// One wave per (b,t,h). Lane owns d = {lane, lane+32, lane+64, lane+96}.

__global__ __launch_bounds__(256) void pack_q(const float* __restrict__ qkvq,
                                              const float* __restrict__ cosb,
                                              const float* __restrict__ sinb,
                                              const float* __restrict__ qw,
                                              bf16_t* __restrict__ Qb) {
  const int lane = threadIdx.x & 31;
  const int idx  = blockIdx.x * 8 + (threadIdx.x >> 5);   // [0, B*T*NH)
  const int h = idx % NH;
  const int t = (idx / NH) % TT;
  const int b = idx / (NH * TT);

  const float* src = qkvq + (size_t)(b * TT + t) * (NH * HD * 2) + h * (HD * 2);
  float x0 = src[lane], x1 = src[lane + 32], x2 = src[lane + 64], x3 = src[lane + 96];
  float ss = wave_sum32(x0 * x0 + x1 * x1 + x2 * x2 + x3 * x3);
  float rr = rsqrtf(ss * (1.0f / 128.0f) + 1e-6f);
  x0 = x0 * rr * qw[lane];      x1 = x1 * rr * qw[lane + 32];
  x2 = x2 * rr * qw[lane + 64]; x3 = x3 * rr * qw[lane + 96];

  const float* cb = cosb + (size_t)(b * TT + t) * HD;
  const float* sb = sinb + (size_t)(b * TT + t) * HD;
  float c0 = cb[lane], c1 = cb[lane + 32], c2 = cb[lane + 64], c3 = cb[lane + 96];
  float s0 = sb[lane], s1 = sb[lane + 32], s2 = sb[lane + 64], s3 = sb[lane + 96];
  float o0 = x0 * c0 - x2 * s0;
  float o1 = x1 * c1 - x3 * s1;
  float o2 = x2 * c2 + x0 * s2;
  float o3 = x3 * c3 + x1 * s3;

  bf16_t* dst = Qb + ((size_t)(b * NH + h) * TT + t) * HD;
  dst[lane] = (bf16_t)o0;       dst[lane + 32] = (bf16_t)o1;
  dst[lane + 64] = (bf16_t)o2;  dst[lane + 96] = (bf16_t)o3;
}

// ---------------------------------------------------------------- pack K (rmsnorm+rope) and V (cast) -> bf16

__global__ __launch_bounds__(256) void pack_kv(const float* __restrict__ qkvk,
                                               const float* __restrict__ qkvv,
                                               const float* __restrict__ cosb,
                                               const float* __restrict__ sinb,
                                               const float* __restrict__ kw,
                                               bf16_t* __restrict__ Kb,
                                               bf16_t* __restrict__ Vb) {
  const int lane = threadIdx.x & 31;
  const int idx  = blockIdx.x * 8 + (threadIdx.x >> 5);   // [0, B*T*NKV)
  const int h = idx % NKV;
  const int t = (idx / NKV) % TT;
  const int b = idx / (NKV * TT);

  const float* src = qkvk + (size_t)(b * TT + t) * (NKV * HD) + h * HD;
  float x0 = src[lane], x1 = src[lane + 32], x2 = src[lane + 64], x3 = src[lane + 96];
  float ss = wave_sum32(x0 * x0 + x1 * x1 + x2 * x2 + x3 * x3);
  float rr = rsqrtf(ss * (1.0f / 128.0f) + 1e-6f);
  x0 = x0 * rr * kw[lane];      x1 = x1 * rr * kw[lane + 32];
  x2 = x2 * rr * kw[lane + 64]; x3 = x3 * rr * kw[lane + 96];

  const float* cb = cosb + (size_t)(b * TT + t) * HD;
  const float* sb = sinb + (size_t)(b * TT + t) * HD;
  float c0 = cb[lane], c1 = cb[lane + 32], c2 = cb[lane + 64], c3 = cb[lane + 96];
  float s0 = sb[lane], s1 = sb[lane + 32], s2 = sb[lane + 64], s3 = sb[lane + 96];

  bf16_t* dk = Kb + ((size_t)(b * NKV + h) * TT + t) * HD;
  dk[lane]      = (bf16_t)(x0 * c0 - x2 * s0);
  dk[lane + 32] = (bf16_t)(x1 * c1 - x3 * s1);
  dk[lane + 64] = (bf16_t)(x2 * c2 + x0 * s2);
  dk[lane + 96] = (bf16_t)(x3 * c3 + x1 * s3);

  const float* vs = qkvv + (size_t)(b * TT + t) * (NKV * HD) + h * HD;
  bf16_t* dv = Vb + ((size_t)(b * NKV + h) * TT + t) * HD;
  dv[lane]      = (bf16_t)vs[lane];
  dv[lane + 32] = (bf16_t)vs[lane + 32];
  dv[lane + 64] = (bf16_t)vs[lane + 64];
  dv[lane + 96] = (bf16_t)vs[lane + 96];
}

// ---------------------------------------------------------------- flash attention
// Block: 128 threads = 4 waves; wave w handles 16 queries [q0base + 16w, +16).
// Key blocks of 32: K staged via TDM (tensor_load_to_lds) when available, V
// transposed into LDS by all threads. Online softmax in f32.

__global__ __launch_bounds__(128) void flash_attn(const bf16_t* __restrict__ Qb,
                                                  const bf16_t* __restrict__ Kb,
                                                  const bf16_t* __restrict__ Vb,
                                                  const float* __restrict__ qkvq,   // for gate
                                                  const int* __restrict__ seg,
                                                  const int* __restrict__ pos,
                                                  float* __restrict__ attn) {
  __shared__ bf16_t lK[32 * 128];     // [key][hd]
  __shared__ bf16_t lV[128 * 32];     // [hd][key]  (V^T)
  __shared__ bf16_t lP[4 * 16 * 32];  // per-wave P staging [row][key]

  const int tid  = threadIdx.x;
  const int wave = tid >> 5, lane = tid & 31;
  const int hf   = lane >> 4, l16 = lane & 15;

  const int bh = blockIdx.x;
  const int b  = bh / NH;
  const int h  = bh % NH;
  const int hkv = h / (NH / NKV);
  const int q0base = blockIdx.y * 64;
  const int q0 = q0base + wave * 16;

  // Q fragments held in registers for the whole KV sweep (4 chunks of K=32 over HD=128)
  const bf16_t* qbase = Qb + ((size_t)(b * NH + h) * TT + q0 + l16) * HD;
  v16bf qf[4];
#pragma unroll
  for (int c = 0; c < 4; ++c) qf[c] = frag_a(qbase + c * 32, hf);

  const v8f vzero = {0.f, 0.f, 0.f, 0.f, 0.f, 0.f, 0.f, 0.f};
  v8f O[8];
#pragma unroll
  for (int t = 0; t < 8; ++t) O[t] = vzero;
  float mrow[8], lrow[8];
  int posq[8], segq[8];
#pragma unroll
  for (int i = 0; i < 8; ++i) {
    mrow[i] = -1e30f; lrow[i] = 0.f;
    int q = q0 + hf * 8 + i;
    posq[i] = pos[b * TT + q];
    segq[i] = seg[b * TT + q];
  }

  const int qmaxw = q0 + 15;
  const int kend  = q0base + 63;

  for (int k0 = 0; k0 <= kend; k0 += 32) {
    __syncthreads();
    const bf16_t* kvsrcK = Kb + ((size_t)(b * NKV + hkv) * TT + k0) * HD;
    const bf16_t* kvsrcV = Vb + ((size_t)(b * NKV + hkv) * TT + k0) * HD;

#if HAVE_TDM
    // K tile (32 x 128 bf16, row-contiguous) -> lK via Tensor Data Mover.
    if (wave == 0) {
      unsigned long long ga = (unsigned long long)(uintptr_t)kvsrcK;
      u32x4 g0;
      g0[0] = 1u;                                            // count=1, user mode
      g0[1] = (unsigned int)(uintptr_t)(void*)lK;            // lds_addr (low 32 = LDS offset)
      g0[2] = (unsigned int)(ga & 0xFFFFFFFFu);              // global_addr[31:0]
      g0[3] = (unsigned int)((ga >> 32) & 0x01FFFFFFu) | 0x80000000u;  // addr[56:32] | type=2
      i32x8 g1;
      g1[0] = 0x00010000;            // workgroup_mask=0, data_size=1 (2 bytes)
      g1[1] = (int)(128u << 16);     // tensor_dim0[15:0]=128 at bits 63:48
      g1[2] = (int)(32u  << 16);     // tensor_dim0 hi=0 | tensor_dim1[15:0]=32
      g1[3] = (int)(128u << 16);     // tensor_dim1 hi=0 | tile_dim0=128
      g1[4] = 32;                    // tile_dim1=32, tile_dim2=0
      g1[5] = 128;                   // tensor_dim0_stride=128
      g1[6] = 0;
      g1[7] = 0;
      i32x4 gz4 = {0, 0, 0, 0};
      i32x8 gz8 = {0, 0, 0, 0, 0, 0, 0, 0};
      // 6-arg form (clang-23 / therock-10.0): (g0, g1, g2, g3, g4, cpol)
      __builtin_amdgcn_tensor_load_to_lds(g0, g1, gz4, gz4, gz8, 0);
    }
    // Meanwhile every thread stages V transposed: 16-byte row loads, b16 scatter.
    {
      v8bf vv[4];
#pragma unroll
      for (int it = 0; it < 4; ++it) {
        int u  = tid + it * 128;       // 512 units of 8 bf16
        int r  = u >> 4;               // key row (32 rows x 16 units)
        int d0 = (u & 15) * 8;
        vv[it] = *(const v8bf*)(kvsrcV + (size_t)r * HD + d0);
      }
#pragma unroll
      for (int it = 0; it < 4; ++it) {
        int u  = tid + it * 128;
        int r  = u >> 4;
        int d0 = (u & 15) * 8;
#pragma unroll
        for (int j = 0; j < 8; ++j) lV[(d0 + j) * 32 + r] = vv[it][j];
      }
    }
    if (wave == 0) __builtin_amdgcn_s_wait_tensorcnt(0);
#else
    {
      v8bf kk[4], vv[4];
#pragma unroll
      for (int it = 0; it < 4; ++it) {
        int u  = tid + it * 128;
        int r  = u >> 4;
        int d0 = (u & 15) * 8;
        kk[it] = *(const v8bf*)(kvsrcK + (size_t)r * HD + d0);
        vv[it] = *(const v8bf*)(kvsrcV + (size_t)r * HD + d0);
      }
#pragma unroll
      for (int it = 0; it < 4; ++it) {
        int u  = tid + it * 128;
        int r  = u >> 4;
        int d0 = (u & 15) * 8;
        *(v8bf*)(lK + r * 128 + d0) = kk[it];
#pragma unroll
        for (int j = 0; j < 8; ++j) lV[(d0 + j) * 32 + r] = vv[it][j];
      }
    }
#endif
    __syncthreads();
    if (k0 > qmaxw) continue;           // wave-uniform: EXEC stays all-ones

    // S = Q K^T  (two 16x16 N-tiles over the 32-key block)
    v8f sacc[2];
#pragma unroll
    for (int nt = 0; nt < 2; ++nt) {
      v8f a = vzero;
#pragma unroll
      for (int c = 0; c < 4; ++c)
        a = wmma_bf16(qf[c], frag_b(lK + (nt * 16 + l16) * 128 + c * 32, hf), a);
      sacc[nt] = a;
    }

    // mask + scale
    float p[2][8];
#pragma unroll
    for (int nt = 0; nt < 2; ++nt) {
      int kk = k0 + nt * 16 + l16;
      int pk = pos[b * TT + kk];
      int sk = seg[b * TT + kk];
#pragma unroll
      for (int i = 0; i < 8; ++i) {
        float v = sacc[nt][i] * SCALE;
        bool ok = (posq[i] >= pk) && (segq[i] == sk);
        p[nt][i] = ok ? v : -1e30f;
      }
    }

    // online softmax (16-lane half reductions: row fixed per half, col = lane&15)
    float alpha[8];
#pragma unroll
    for (int i = 0; i < 8; ++i) {
      float v = fmaxf(p[0][i], p[1][i]);
#pragma unroll
      for (int off = 8; off > 0; off >>= 1) v = fmaxf(v, __shfl_xor(v, off, 32));
      float mnew = fmaxf(mrow[i], v);
      alpha[i] = __expf(mrow[i] - mnew);
      float e0 = __expf(p[0][i] - mnew);
      float e1 = __expf(p[1][i] - mnew);
      p[0][i] = e0; p[1][i] = e1;
      float rs = e0 + e1;
#pragma unroll
      for (int off = 8; off > 0; off >>= 1) rs += __shfl_xor(rs, off, 32);
      lrow[i] = lrow[i] * alpha[i] + rs;
      mrow[i] = mnew;
    }
#pragma unroll
    for (int t = 0; t < 8; ++t)
#pragma unroll
      for (int i = 0; i < 8; ++i) O[t][i] *= alpha[i];

    // C-layout -> A-layout for P via per-wave LDS bounce
    bf16_t* lPw = lP + wave * 512;
#pragma unroll
    for (int nt = 0; nt < 2; ++nt)
#pragma unroll
      for (int i = 0; i < 8; ++i)
        lPw[(hf * 8 + i) * 32 + nt * 16 + l16] = (bf16_t)p[nt][i];
    asm volatile("s_wait_dscnt 0" ::: "memory");   // same-wave LDS RAW

    v16bf pf = frag_a(lPw + l16 * 32, hf);
#pragma unroll
    for (int nt2 = 0; nt2 < 8; ++nt2)
      O[nt2] = wmma_bf16(pf, frag_b(lV + (nt2 * 16 + l16) * 32, hf), O[nt2]);
  }

  // epilogue: 1/l normalization, sigmoid gate, write fp32 attn [B*T, NH*HD]
#pragma unroll
  for (int nt2 = 0; nt2 < 8; ++nt2) {
#pragma unroll
    for (int i = 0; i < 8; ++i) {
      int q = q0 + hf * 8 + i;
      int d = nt2 * 16 + l16;
      float g = qkvq[(size_t)(b * TT + q) * (NH * HD * 2) + h * (HD * 2) + HD + d];
      float val = O[nt2][i] * (1.0f / lrow[i]) * (1.0f / (1.0f + __expf(-g)));
      attn[(size_t)(b * TT + q) * (NH * HD) + h * HD + d] = val;
    }
  }
}

// ---------------------------------------------------------------- launch

extern "C" void kernel_launch(void* const* d_in, const int* in_sizes, int n_in,
                              void* d_out, int out_size, void* d_ws, size_t ws_size,
                              hipStream_t stream) {
  const float* x    = (const float*)d_in[0];
  const float* cosb = (const float*)d_in[1];
  const float* sinb = (const float*)d_in[2];
  const float* wq   = (const float*)d_in[3];
  const float* wk   = (const float*)d_in[4];
  const float* wv   = (const float*)d_in[5];
  const float* wo   = (const float*)d_in[6];
  const float* qnw  = (const float*)d_in[7];
  const float* knw  = (const float*)d_in[8];
  const int*   seg  = (const int*)d_in[9];
  const int*   pos  = (const int*)d_in[10];
  float* out = (float*)d_out;

  char* ws = (char*)d_ws;
  float*  qkvq = (float*)(ws);                      // [4096, 4096] fp32
  float*  qkvk = (float*)(ws + 67108864);           // [4096, 512]  fp32
  float*  qkvv = (float*)(ws + 75497472);           // [4096, 512]  fp32
  bf16_t* Qb   = (bf16_t*)(ws + 83886080);          // [B,NH,T,HD]  bf16
  bf16_t* Kb   = (bf16_t*)(ws + 100663296);         // [B,NKV,T,HD] bf16
  bf16_t* Vb   = (bf16_t*)(ws + 104857600);         // [B,NKV,T,HD] bf16
  float*  attn = (float*)(ws + 109051904);          // [4096, 2048] fp32

  // QKV projections (bf16 WMMA)
  gemm_bf16w<<<dim3(64, 64), 128, 0, stream>>>(x, wq, qkvq, 4096, 4096, 2048);
  gemm_bf16w<<<dim3(64, 8),  128, 0, stream>>>(x, wk, qkvk, 4096, 512,  2048);
  gemm_bf16w<<<dim3(64, 8),  128, 0, stream>>>(x, wv, qkvv, 4096, 512,  2048);

  // norm + rope + bf16 pack
  pack_q <<<8192, 256, 0, stream>>>(qkvq, cosb, sinb, qnw, Qb);
  pack_kv<<<2048, 256, 0, stream>>>(qkvk, qkvv, cosb, sinb, knw, Kb, Vb);

  // flash attention + gate
  flash_attn<<<dim3(BB * NH, TT / 64), 128, 0, stream>>>(Qb, Kb, Vb, qkvq, seg, pos, attn);

  // output projection
  gemm_bf16w<<<dim3(64, 32), 128, 0, stream>>>(attn, wo, out, 4096, 2048, 2048);
}
